// AttentiveReader_64931315581665
// MI455X (gfx1250) — matrix operand order repaired and checked by reading, not compiled, not run on hardware
//
#include <hip/hip_runtime.h>
#include <hip/hip_bf16.h>

// ---------------------------------------------------------------------------
// AttentiveReader on gfx1250 (MI455X)
//   H=128, B=32, TD=1000, TQ=50, V=264588
// ---------------------------------------------------------------------------

typedef __attribute__((ext_vector_type(16))) __bf16 v16bf;
typedef __attribute__((ext_vector_type(8)))  __bf16 v8bf;
typedef __attribute__((ext_vector_type(8)))  float  v8f;
typedef __attribute__((ext_vector_type(2)))  float  v2f;

#define HH   128
#define BB   32
#define TDn  1000
#define TQn  50
#define VV   264588
#define H2   256
#define H4   512

__device__ __forceinline__ float sigf(float x) { return 1.0f / (1.0f + __expf(-x)); }

// ---------------------------------------------------------------------------
// K1: embedding gather, time-major.  grid = TD*B + TQ*B rows, 128 thr/row.
// ---------------------------------------------------------------------------
__global__ __launch_bounds__(128) void embed_kernel(
    const int* __restrict__ doc, const int* __restrict__ qry,
    const float* __restrict__ emb, float* __restrict__ ed, float* __restrict__ eq) {
  int row = blockIdx.x;
  int h = threadIdx.x;
  if (row < TDn * BB) {
    int t = row >> 5, b = row & 31;
    int tok = doc[b * TDn + t];
    ed[(size_t)row * HH + h] = emb[(size_t)tok * HH + h];
  } else {
    int row2 = row - TDn * BB;
    int t = row2 >> 5, b = row2 & 31;
    int tok = qry[b * TQn + t];
    eq[(size_t)row2 * HH + h] = emb[(size_t)tok * HH + h];
  }
}

// ---------------------------------------------------------------------------
// K2: persistent bf16-WMMA LSTM scan. grid = 4 (doc fw/bw, query fw/bw).
// Weights bf16, pre-swizzled into LDS in WMMA B-fragment order (256 KB).
// Each wave owns 16 columns of all 4 gates; c state stays in registers.
// ---------------------------------------------------------------------------
__global__ __launch_bounds__(256) void lstm_kernel(
    const float* __restrict__ ed, const float* __restrict__ eq,
    const float* __restrict__ Wdf, const float* __restrict__ bdf,
    const float* __restrict__ Wdb, const float* __restrict__ bdb,
    const float* __restrict__ Wqf, const float* __restrict__ bqf,
    const float* __restrict__ Wqb, const float* __restrict__ bqb,
    float* __restrict__ dfw, float* __restrict__ dbw,
    float* __restrict__ qfw, float* __restrict__ qbw) {
  __shared__ __align__(32) __bf16 Wl[H2 * H4];   // 256 KB, fragment-ordered
  __shared__ __align__(16) __bf16 xl[BB * HH];   // 8 KB
  __shared__ __align__(16) __bf16 hl[BB * HH];   // 8 KB

  const float* X; const float* W; const float* bias; float* out; int T; int rev;
  switch (blockIdx.x) {
    case 0:  X = ed; W = Wdf; bias = bdf; out = dfw; T = TDn; rev = 0; break;
    case 1:  X = ed; W = Wdb; bias = bdb; out = dbw; T = TDn; rev = 1; break;
    case 2:  X = eq; W = Wqf; bias = bqf; out = qfw; T = TQn; rev = 0; break;
    default: X = eq; W = Wqb; bias = bqb; out = qbw; T = TQn; rev = 1; break;
  }
  int tid = threadIdx.x, wave = tid >> 5, lane = tid & 31;
  int laneHalf = lane >> 4, lane16 = lane & 15;

  // Swizzle W (f32 [256][512]) -> bf16 fragments:
  // frag f=(ktile*32+ntile); lane l; elem j:  K=ktile*32+16*(l>>4)+j, N=ntile*16+(l&15)
  for (int e = tid; e < H2 * H4; e += 256) {
    int f = e >> 9, rem = e & 511;
    int l = rem >> 4, j = rem & 15;
    int ktile = f >> 5, ntile = f & 31;
    int K = ktile * 32 + ((l >> 4) << 4) + j;
    int N = ntile * 16 + (l & 15);
    Wl[e] = (__bf16)W[K * H4 + N];
  }
  for (int e = tid; e < BB * HH; e += 256) hl[e] = (__bf16)0.0f;

  int col = wave * 16 + lane16;              // output column (0..127)
  float bi = bias[col], bj = bias[HH + col];
  float bff = bias[2 * HH + col], bo = bias[3 * HH + col];

  v8f c0 = {}, c1 = {};                      // cell state, accumulator layout
  for (int step = 0; step < T; ++step) {
    int t = rev ? (T - 1 - step) : step;
    const float* xp = X + (size_t)t * (BB * HH);
    for (int e = tid; e < BB * HH; e += 256) xl[e] = (__bf16)xp[e];
    __syncthreads();                         // xl staged, prev hl visible

    v8f z0[4] = {v8f{}, v8f{}, v8f{}, v8f{}};
    v8f z1[4] = {v8f{}, v8f{}, v8f{}, v8f{}};
#pragma unroll
    for (int k0 = 0; k0 < 8; ++k0) {
      const __bf16* src = (k0 < 4) ? xl : hl;
      int kb = ((k0 & 3) * 32) + 16 * laneHalf;  // per-lane K base
      v16bf A0, A1;
      const __bf16* p0 = src + lane16 * HH + kb;
      const __bf16* p1 = src + (16 + lane16) * HH + kb;
      ((v8bf*)&A0)[0] = *(const v8bf*)p0;
      ((v8bf*)&A0)[1] = *(const v8bf*)(p0 + 8);
      ((v8bf*)&A1)[0] = *(const v8bf*)p1;
      ((v8bf*)&A1)[1] = *(const v8bf*)(p1 + 8);
#pragma unroll
      for (int gate = 0; gate < 4; ++gate) {
        int ntile = gate * 8 + wave;
        v16bf Bf = *(const v16bf*)(Wl + ((k0 * 32 + ntile) << 9) + lane * 16);
        z0[gate] = __builtin_amdgcn_wmma_f32_16x16x32_bf16(
            false, A0, false, Bf, (short)0, z0[gate], false, false);
        z1[gate] = __builtin_amdgcn_wmma_f32_16x16x32_bf16(
            false, A1, false, Bf, (short)0, z1[gate], false, false);
      }
    }
    __syncthreads();                         // all hl/xl reads done
#pragma unroll
    for (int rr = 0; rr < 8; ++rr) {
      int m = rr + 8 * laneHalf;
      float i0 = sigf(z0[0][rr] + bi);
      float g0 = tanhf(z0[1][rr] + bj);
      float f0 = sigf(z0[2][rr] + bff);
      float o0 = sigf(z0[3][rr] + bo);
      c0[rr] = c0[rr] * f0 + i0 * g0;
      float h0 = tanhf(c0[rr]) * o0;
      hl[m * HH + col] = (__bf16)h0;
      out[(size_t)t * (BB * HH) + m * HH + col] = h0;

      float i1 = sigf(z1[0][rr] + bi);
      float g1 = tanhf(z1[1][rr] + bj);
      float f1 = sigf(z1[2][rr] + bff);
      float o1 = sigf(z1[3][rr] + bo);
      c1[rr] = c1[rr] * f1 + i1 * g1;
      float h1 = tanhf(c1[rr]) * o1;
      hl[(16 + m) * HH + col] = (__bf16)h1;
      out[(size_t)t * (BB * HH) + (16 + m) * HH + col] = h1;
    }
  }
}

// ---------------------------------------------------------------------------
// K3: u assembly + tanh attention + softmax + pooled r.  grid = B blocks.
// ---------------------------------------------------------------------------
__global__ __launch_bounds__(256) void attn_kernel(
    const float* __restrict__ dfw, const float* __restrict__ dbw,
    const float* __restrict__ qfw, const float* __restrict__ qbw,
    const float* __restrict__ W_ym, const float* __restrict__ W_um,
    float* __restrict__ u_out, float* __restrict__ r_out) {
  __shared__ float wy[H2];
  __shared__ float sc[TDn];
  __shared__ float red[256];
  int b = blockIdx.x, tid = threadIdx.x;

  float uval = (tid < HH) ? qfw[((size_t)(TQn - 1) * BB + b) * HH + tid]
                          : qbw[(size_t)b * HH + (tid - HH)];
  wy[tid] = W_ym[tid];
  u_out[b * H2 + tid] = uval;
  red[tid] = uval * W_um[tid];
  __syncthreads();
  for (int s = 128; s > 0; s >>= 1) {
    if (tid < s) red[tid] += red[tid + s];
    __syncthreads();
  }
  float udot = red[0];
  __syncthreads();

  float lmax = -1e30f;
  for (int t = tid; t < TDn; t += 256) {
    const float* pf = dfw + ((size_t)t * BB + b) * HH;
    const float* pb = dbw + ((size_t)t * BB + b) * HH;
    float dot = udot;
    for (int h = 0; h < HH; ++h) dot += pf[h] * wy[h] + pb[h] * wy[HH + h];
    float m = tanhf(dot);
    sc[t] = m;
    lmax = fmaxf(lmax, m);
  }
  red[tid] = lmax;
  __syncthreads();
  for (int s = 128; s > 0; s >>= 1) {
    if (tid < s) red[tid] = fmaxf(red[tid], red[tid + s]);
    __syncthreads();
  }
  float gmax = red[0];
  __syncthreads();
  float lsum = 0.f;
  for (int t = tid; t < TDn; t += 256) {
    float e = __expf(sc[t] - gmax);
    sc[t] = e;
    lsum += e;
  }
  red[tid] = lsum;
  __syncthreads();
  for (int s = 128; s > 0; s >>= 1) {
    if (tid < s) red[tid] += red[tid + s];
    __syncthreads();
  }
  float inv = 1.0f / red[0];
  __syncthreads();

  float acc = 0.f;
  if (tid < HH) {
    for (int t = 0; t < TDn; ++t) acc += sc[t] * dfw[((size_t)t * BB + b) * HH + tid];
  } else {
    for (int t = 0; t < TDn; ++t) acc += sc[t] * dbw[((size_t)t * BB + b) * HH + (tid - HH)];
  }
  r_out[b * H2 + tid] = acc * inv;
}

// ---------------------------------------------------------------------------
// K4: g = relu([r|u] @ [W_rg;W_ug]) via f32 WMMA 16x16x4 — HBM-bound stream.
// One wave per 16-column N-tile; NT loads keep the 542 MB stream out of L2.
// ---------------------------------------------------------------------------
__global__ __launch_bounds__(256) void outgemm_kernel(
    const float* __restrict__ rbuf, const float* __restrict__ ubuf,
    const float* __restrict__ W_rg, const float* __restrict__ W_ug,
    float* __restrict__ g) {
  __shared__ float As[BB * H4];  // 64 KB: A' = [r | u]
  int tid = threadIdx.x;
  for (int e = tid; e < BB * H4; e += 256) {
    int m = e >> 9, k = e & 511;
    As[e] = (k < H2) ? rbuf[m * H2 + k] : ubuf[m * H2 + (k - H2)];
  }
  __syncthreads();

  int wave = tid >> 5, lane = tid & 31, laneHalf = lane >> 4, lane16 = lane & 15;
  long nt = (long)blockIdx.x * 8 + wave;
  long n = nt * 16;
  long nc = n + lane16; if (nc >= VV) nc = VV - 1;   // clamped loads, guarded stores

  v8f acc0 = {}, acc1 = {};
#pragma unroll 4
  for (int k0 = 0; k0 < H4; k0 += 4) {
    int kk = k0 + 2 * laneHalf;
    v2f A0 = *(const v2f*)(As + lane16 * H4 + kk);
    v2f A1 = *(const v2f*)(As + (16 + lane16) * H4 + kk);
    const float* Bbase = (k0 < H2) ? W_rg : W_ug;
    long koff = (long)(kk & 255);
    float b0 = __builtin_nontemporal_load(Bbase + koff * (long)VV + nc);
    float b1 = __builtin_nontemporal_load(Bbase + (koff + 1) * (long)VV + nc);
    v2f Bf = {b0, b1};
    acc0 = __builtin_amdgcn_wmma_f32_16x16x4_f32(
        false, A0, false, Bf, (short)0, acc0, false, false);
    acc1 = __builtin_amdgcn_wmma_f32_16x16x4_f32(
        false, A1, false, Bf, (short)0, acc1, false, false);
    int kp = k0 + 32;                         // prefetch ~2 KB ahead in the stream
    if (kp < H4) {
      const float* Pb = (kp < H2) ? W_rg : W_ug;
      long kpo = (long)((kp + 2 * laneHalf) & 255);
      __builtin_prefetch(Pb + kpo * (long)VV + nc, 0, 1);
    }
  }
  long vcol = n + lane16;
  if (vcol < VV) {
#pragma unroll
    for (int rr = 0; rr < 8; ++rr) {
      int m0 = rr + 8 * laneHalf;
      g[(size_t)m0 * VV + vcol]        = fmaxf(acc0[rr], 0.0f);
      g[(size_t)(16 + m0) * VV + vcol] = fmaxf(acc1[rr], 0.0f);
    }
  }
}

// ---------------------------------------------------------------------------
extern "C" void kernel_launch(void* const* d_in, const int* in_sizes, int n_in,
                              void* d_out, int out_size, void* d_ws, size_t ws_size,
                              hipStream_t stream) {
  const int*   document = (const int*)d_in[0];
  const int*   query    = (const int*)d_in[1];
  const float* emb      = (const float*)d_in[2];
  const float* Wd_fw    = (const float*)d_in[3];
  const float* bd_fw    = (const float*)d_in[4];
  const float* Wd_bw    = (const float*)d_in[5];
  const float* bd_bw    = (const float*)d_in[6];
  const float* Wq_fw    = (const float*)d_in[7];
  const float* bq_fw    = (const float*)d_in[8];
  const float* Wq_bw    = (const float*)d_in[9];
  const float* bq_bw    = (const float*)d_in[10];
  const float* W_ym     = (const float*)d_in[11];
  const float* W_um     = (const float*)d_in[12];
  const float* W_rg     = (const float*)d_in[13];
  const float* W_ug     = (const float*)d_in[14];
  float* g = (float*)d_out;

  // Workspace carve-up (floats): ~51.7 MB total
  float* ws  = (float*)d_ws;
  float* ed  = ws;                                   // TD*B*H = 4,096,000
  float* eq  = ed  + (size_t)TDn * BB * HH;          // TQ*B*H =   204,800
  float* dfw = eq  + (size_t)TQn * BB * HH;
  float* dbw = dfw + (size_t)TDn * BB * HH;
  float* qfw = dbw + (size_t)TDn * BB * HH;
  float* qbw = qfw + (size_t)TQn * BB * HH;
  float* u   = qbw + (size_t)TQn * BB * HH;          // B*2H = 8192
  float* r   = u   + (size_t)BB * H2;                // B*2H = 8192

  embed_kernel<<<TDn * BB + TQn * BB, 128, 0, stream>>>(document, query, emb, ed, eq);

  lstm_kernel<<<4, 256, 0, stream>>>(ed, eq,
                                     Wd_fw, bd_fw, Wd_bw, bd_bw,
                                     Wq_fw, bq_fw, Wq_bw, bq_bw,
                                     dfw, dbw, qfw, qbw);

  attn_kernel<<<BB, 256, 0, stream>>>(dfw, dbw, qfw, qbw, W_ym, W_um, u, r);

  int ntiles = (VV + 15) / 16;          // 16537
  int blocks = (ntiles + 7) / 8;        // 2068
  outgemm_kernel<<<blocks, 256, 0, stream>>>(r, u, W_rg, W_ug, g);
}
